// deformable_LKA_75136157877131
// MI455X (gfx1250) — matrix operand order repaired and checked by reading, not compiled, not run on hardware
//
#include <hip/hip_runtime.h>

// ---------------------------------------------------------------------------
// Deformable LKA for MI455X (gfx1250, wave32).
//  * GEMM stages (offset convs, 1x1 convs): v_wmma_f32_16x16x32_bf16 with a
//    bf16 hi/lo split (3 WMMAs per 32-wide K chunk) for ~fp32 accuracy.
//  * 1x1 conv stages the 64x16 f32 activation tile into LDS via the Tensor
//    Data Mover (tensor_load_to_lds + s_wait_tensorcnt), WMMA B-fragments are
//    then built from LDS with ds_load. Fallback: cooperative LDS staging.
//  * Deformable bilinear sampling: scalar VALU + gathers (no matrix shape).
// ---------------------------------------------------------------------------

typedef __attribute__((ext_vector_type(16))) __bf16        v16bf;
typedef __attribute__((ext_vector_type(8)))  float         v8f;
typedef __attribute__((ext_vector_type(4)))  unsigned int  u32x4;
typedef __attribute__((ext_vector_type(4)))  int           i32x4;
typedef __attribute__((ext_vector_type(8)))  int           i32x8;

#define IMG_H   64
#define IMG_W   64
#define HWSZ    4096          // 64*64
#define CIN     64            // channels per branch
#define NBATCH  2

// TDM availability (device pass only; host pass parses the fallback).
#if defined(__HIP_DEVICE_COMPILE__) && defined(__has_builtin)
#if __has_builtin(__builtin_amdgcn_tensor_load_to_lds) && \
    __has_builtin(__builtin_amdgcn_s_wait_tensorcnt)
#define HAVE_TDM 1
#endif
#endif
#ifndef HAVE_TDM
#define HAVE_TDM 0
#endif

__device__ __forceinline__ void bf16_split(float v, __bf16& hi, __bf16& lo) {
  __bf16 h = (__bf16)v;            // RNE conversion to bf16
  hi = h;
  lo = (__bf16)(v - (float)h);     // residual, recovers ~16 more mantissa bits
}

// ---------------------------------------------------------------------------
// Offset convolution as implicit GEMM:
//   D[m][n] = sum_K  W[m][K] * im2col(x)[K][n],   K = cin*k*k + ky*k + kx
// One wave per 16(M-chan) x 16(N-pixel) tile. Ktot = 64*k*k (multiple of 32).
// Weight row m is contiguous in K (OIHW flattening == row-major K).
// ---------------------------------------------------------------------------
__global__ void __launch_bounds__(32)
k_offconv_wmma(const float* __restrict__ yin, int inBatchStride,
               const float* __restrict__ wgt, const float* __restrict__ bias,
               float* __restrict__ out, int Cout, int ksz, int pad, int dil)
{
  const int lane = threadIdx.x;        // 0..31 (wave32)
  const int h16  = lane >> 4;          // which half-wave
  const int ln   = lane & 15;

  const int tile_n = blockIdx.x;       // 0..255  (4096 pixels / 16)
  const int tile_m = blockIdx.y;       // ceil(Cout/16) tiles
  const int b      = blockIdx.z;

  const int kk   = ksz * ksz;
  const int Ktot = CIN * kk;

  const int m  = tile_m * 16 + ln;     // A row (output channel) for this lane
  const int n  = tile_n * 16 + ln;     // B/D column (pixel) for this lane
  const int ph = n >> 6;               // pixel y
  const int pw = n & 63;               // pixel x

  const float* src  = yin + (long)b * inBatchStride;
  const float* wrow = wgt + (long)m * Ktot;      // contiguous A row

  v8f acc = {0.f, 0.f, 0.f, 0.f, 0.f, 0.f, 0.f, 0.f};

  for (int kb = 0; kb < Ktot; kb += 32) {
    v16bf ahi, alo, bhi, blo;

    if (m < Cout && kb + 64 <= Ktot)
      __builtin_prefetch(wrow + kb + 32, 0, 1);  // global_prefetch_b8

    // A fragment (weights), contiguous k order, same as B:
    //   K = kb + e + 16*h16
#pragma unroll
    for (int e = 0; e < 16; ++e) {
      int kidx = kb + e + 16 * h16;
      float v = (m < Cout) ? wrow[kidx] : 0.f;
      __bf16 hb, lb; bf16_split(v, hb, lb);
      ahi[e] = hb; alo[e] = lb;
    }

    // B fragment (im2col in registers), B column-striped layout:
    //   K = kb + e + 16*h16, column n
#pragma unroll
    for (int e = 0; e < 16; ++e) {
      int kidx = kb + e + 16 * h16;
      int cin = kidx / kk;
      int r   = kidx - cin * kk;
      int ky  = r / ksz;
      int kx  = r - ky * ksz;
      int iy  = ph + ky * dil - pad;
      int ix  = pw + kx * dil - pad;
      float v = 0.f;
      if (iy >= 0 && iy < IMG_H && ix >= 0 && ix < IMG_W)
        v = src[(cin << 12) + (iy << 6) + ix];
      __bf16 hb, lb; bf16_split(v, hb, lb);
      bhi[e] = hb; blo[e] = lb;
    }

    // hi*hi + hi*lo + lo*hi  (lo*lo term ~2^-32, dropped)
    acc = __builtin_amdgcn_wmma_f32_16x16x32_bf16(false, ahi, false, bhi, (short)0, acc, false, false);
    acc = __builtin_amdgcn_wmma_f32_16x16x32_bf16(false, ahi, false, blo, (short)0, acc, false, false);
    acc = __builtin_amdgcn_wmma_f32_16x16x32_bf16(false, alo, false, bhi, (short)0, acc, false, false);
  }

  // D layout: lane column n, VGPR r -> row r + 8*h16
  float* dst = out + (long)b * Cout * HWSZ;
#pragma unroll
  for (int r = 0; r < 8; ++r) {
    int mm = tile_m * 16 + r + 8 * h16;
    if (mm < Cout)
      dst[(long)mm * HWSZ + n] = acc[r] + bias[mm];
  }
}

// ---------------------------------------------------------------------------
// Deformable depthwise bilinear sampling + reduce over k*k points.
// One thread per (b, c, h, w). Offsets shared across channels (L2-friendly).
// ---------------------------------------------------------------------------
__global__ void k_deform(const float* __restrict__ yin, int inBatchStride,
                         const float* __restrict__ off,   // (B, 2K, H, W)
                         const float* __restrict__ dww,   // (C, K)
                         float* __restrict__ out,         // (B, C, H, W)
                         int ksz, int pad, int dil)
{
  int idx = blockIdx.x * blockDim.x + threadIdx.x;
  if (idx >= NBATCH * CIN * HWSZ) return;

  const int K   = ksz * ksz;
  const int b   = idx >> 18;              // / (64*4096)
  const int rem = idx & ((1 << 18) - 1);
  const int c   = rem >> 12;              // / 4096
  const int pix = rem & 4095;
  const int h   = pix >> 6;
  const int w   = pix & 63;

  const float* src  = yin + (long)b * inBatchStride + (c << 12);
  const float* offb = off + (long)b * (2 * K) * HWSZ + pix;

  float acc = 0.f;
  for (int kv = 0; kv < K; ++kv) {
    float oy = offb[(long)(2 * kv)     * HWSZ];
    float ox = offb[(long)(2 * kv + 1) * HWSZ];
    float py = oy + (float)(h - pad + (kv / ksz) * dil);
    float px = ox + (float)(w - pad + (kv % ksz) * dil);
    float y0f = floorf(py), x0f = floorf(px);
    float ly = py - y0f, lx = px - x0f;
    int y0 = (int)y0f, x0 = (int)x0f;

    float v = 0.f;
#pragma unroll
    for (int cy = 0; cy < 2; ++cy) {
#pragma unroll
      for (int cx = 0; cx < 2; ++cx) {
        int yi = y0 + cy, xi = x0 + cx;
        float wy = cy ? ly : 1.f - ly;
        float wx = cx ? lx : 1.f - lx;
        if (yi >= 0 && yi < IMG_H && xi >= 0 && xi < IMG_W)
          v += src[(yi << 6) + xi] * wy * wx;
      }
    }
    acc += v * dww[c * K + kv];
  }
  out[((long)b * CIN + c) * HWSZ + pix] = acc;
}

// ---------------------------------------------------------------------------
// 1x1 conv (M=64, K=64 GEMM per 16-pixel tile) fused with bias and the final
// elementwise  out = x * a  for one 64-channel half of the output.
// The 64(K) x 16(N) f32 activation tile is staged to LDS by the Tensor Data
// Mover (2D D#: tile 16x64, stride 4096 elem, data_size 4B), then B-fragments
// are built from LDS with ds_load_b32.
// ---------------------------------------------------------------------------
__global__ void __launch_bounds__(32)
k_conv1x1_mul(const float* __restrict__ act,   // (B,64,H,W), batch stride 64*HW
              const float* __restrict__ wgt,   // (64,64)
              const float* __restrict__ bias,  // (64)
              const float* __restrict__ xfull, // (B,128,H,W)
              float* __restrict__ outp,        // (B,128,H,W)
              int chOff)                       // 0 or 64
{
  __shared__ float tileB[CIN * 16];            // K x N tile, 4 KB

  const int lane = threadIdx.x;
  const int h16  = lane >> 4;
  const int ln   = lane & 15;
  const int tile_n = blockIdx.x;   // 0..255
  const int tile_m = blockIdx.y;   // 0..3
  const int b      = blockIdx.z;

  const int m  = tile_m * 16 + ln;
  const int n0 = tile_n * 16;
  const int n  = n0 + ln;

  const float* src = act + (long)b * CIN * HWSZ + n0;   // tile base (row K=0)

#if HAVE_TDM
  {
    // ---- Tensor DMA descriptor (CDNA5 ISA) ----
    unsigned long long ga = (unsigned long long)(const void*)src;
    unsigned ldsoff = (unsigned)(unsigned long long)(void*)tileB; // addr[31:0] = LDS offset

    u32x4 g0;
    g0[0] = 1u;                                      // count=1 (valid), user mode
    g0[1] = ldsoff;                                  // lds_addr [63:32]
    g0[2] = (unsigned)(ga & 0xffffffffull);          // global_addr lo
    g0[3] = (unsigned)((ga >> 32) & 0x1ffffffull)    // global_addr[56:32]
          | (2u << 30);                              // type=2 ("image")

    i32x8 g1;
    g1[0] = (int)(2u << 16);     // wg_mask=0 | data_size=2 (4 bytes), no flags
    g1[1] = (int)(16u << 16);    // atomic_barrier_addr=0 | tensor_dim0=16 (lo16)
    g1[2] = (int)(64u << 16);    // tensor_dim0 hi=0 | tensor_dim1=64 (lo16)
    g1[3] = (int)(16u << 16);    // tensor_dim1 hi=0 | tile_dim0=16
    g1[4] = 64;                  // tile_dim1=64 | tile_dim2=0
    g1[5] = 4096;                // tensor_dim0_stride lo32 (elements)
    g1[6] = 0;                   // stride0 hi | stride1 lo16
    g1[7] = 0;                   // stride1 hi

    i32x4 g2 = {0, 0, 0, 0};     // 2D tile: dims 2/3 unused
    i32x4 g3 = {0, 0, 0, 0};

#if __clang_major__ >= 23
    i32x8 g1z = {0, 0, 0, 0, 0, 0, 0, 0};
    __builtin_amdgcn_tensor_load_to_lds(g0, g1, g2, g3, g1z, 0);
#else
    __builtin_amdgcn_tensor_load_to_lds(g0, g1, g2, g3, 0);
#endif
    __builtin_amdgcn_s_wait_tensorcnt(0);            // TENSORcnt == 0
    // TDM wrote LDS behind the compiler's back: full memory barrier so the
    // ds_loads below are neither folded to undef nor hoisted above the wait.
    __asm__ __volatile__("" ::: "memory");
  }
#else
  // Cooperative fallback: 1024 floats / 32 lanes = 32 each.
  for (int t = lane; t < CIN * 16; t += 32)
    tileB[t] = src[(long)(t >> 4) * HWSZ + (t & 15)];
  __syncthreads();
#endif

  v8f acc = {0.f, 0.f, 0.f, 0.f, 0.f, 0.f, 0.f, 0.f};

  for (int kb = 0; kb < CIN; kb += 32) {
    v16bf ahi, alo, bhi, blo;
#pragma unroll
    for (int e = 0; e < 16; ++e) {
      int kidx = kb + e + 16 * h16;
      __bf16 hb, lb; bf16_split(wgt[m * CIN + kidx], hb, lb);
      ahi[e] = hb; alo[e] = lb;
    }
#pragma unroll
    for (int e = 0; e < 16; ++e) {
      int kidx = kb + e + 16 * h16;
      __bf16 hb, lb; bf16_split(tileB[kidx * 16 + ln], hb, lb);  // ds_load_b32
      bhi[e] = hb; blo[e] = lb;
    }
    acc = __builtin_amdgcn_wmma_f32_16x16x32_bf16(false, ahi, false, bhi, (short)0, acc, false, false);
    acc = __builtin_amdgcn_wmma_f32_16x16x32_bf16(false, ahi, false, blo, (short)0, acc, false, false);
    acc = __builtin_amdgcn_wmma_f32_16x16x32_bf16(false, alo, false, bhi, (short)0, acc, false, false);
  }

#pragma unroll
  for (int r = 0; r < 8; ++r) {
    int mm = tile_m * 16 + r + 8 * h16;
    long o = ((long)b * 128 + chOff + mm) * HWSZ + n;
    outp[o] = xfull[o] * (acc[r] + bias[mm]);
  }
}

// ---------------------------------------------------------------------------
extern "C" void kernel_launch(void* const* d_in, const int* in_sizes, int n_in,
                              void* d_out, int out_size, void* d_ws, size_t ws_size,
                              hipStream_t stream) {
  (void)in_sizes; (void)n_in; (void)out_size; (void)ws_size;

  const float* x         = (const float*)d_in[0];   // (2,128,64,64)
  const float* cv0_off_w = (const float*)d_in[1];   // (18,64,3,3)
  const float* cv0_off_b = (const float*)d_in[2];
  const float* cv0_w     = (const float*)d_in[3];   // (64,1,3,3)
  const float* cvs_off_w = (const float*)d_in[4];   // (50,64,5,5)
  const float* cvs_off_b = (const float*)d_in[5];
  const float* cvs_w     = (const float*)d_in[6];   // (64,1,5,5)
  const float* c0_off_w  = (const float*)d_in[7];   // (50,64,5,5)
  const float* c0_off_b  = (const float*)d_in[8];
  const float* c0_w      = (const float*)d_in[9];   // (64,1,5,5)
  const float* cs_off_w  = (const float*)d_in[10];  // (98,64,7,7)
  const float* cs_off_b  = (const float*)d_in[11];
  const float* cs_w      = (const float*)d_in[12];  // (64,1,7,7)
  const float* conv1_w   = (const float*)d_in[13];  // (64,64,1,1)
  const float* conv1_b   = (const float*)d_in[14];
  const float* conv2_w   = (const float*)d_in[15];
  const float* conv2_b   = (const float*)d_in[16];
  float* out = (float*)d_out;

  // Workspace: bufA | bufB (2 MB each) | offBuf (3.14 MB) => ~7.1 MB total
  float* ws     = (float*)d_ws;
  float* bufA   = ws;                         // 2*64*4096 = 524288 f
  float* bufB   = ws + 524288;                // 524288 f
  float* offBuf = ws + 1048576;               // up to 2*98*4096 = 802816 f

  const int xStride = 128 * HWSZ;             // batch stride inside x
  const int aStride = 64 * HWSZ;              // batch stride of packed buffers
  dim3 blk32(32, 1, 1);
  dim3 blkD(256, 1, 1);
  dim3 grdD(2048, 1, 1);                      // 2*64*4096 threads

  // ---- Branch 0 (channels 0..63) ----
  k_offconv_wmma<<<dim3(256, 2, NBATCH), blk32, 0, stream>>>(
      x, xStride, cv0_off_w, cv0_off_b, offBuf, 18, 3, 1, 1);
  k_deform<<<grdD, blkD, 0, stream>>>(x, xStride, offBuf, cv0_w, bufA, 3, 1, 1);

  k_offconv_wmma<<<dim3(256, 4, NBATCH), blk32, 0, stream>>>(
      bufA, aStride, cvs_off_w, cvs_off_b, offBuf, 50, 5, 6, 3);
  k_deform<<<grdD, blkD, 0, stream>>>(bufA, aStride, offBuf, cvs_w, bufB, 5, 6, 3);

  k_conv1x1_mul<<<dim3(256, 4, NBATCH), blk32, 0, stream>>>(
      bufB, conv1_w, conv1_b, x, out, 0);

  // ---- Branch 1 (channels 64..127) ----
  k_offconv_wmma<<<dim3(256, 4, NBATCH), blk32, 0, stream>>>(
      x + 64 * HWSZ, xStride, c0_off_w, c0_off_b, offBuf, 50, 5, 2, 1);
  k_deform<<<grdD, blkD, 0, stream>>>(x + 64 * HWSZ, xStride, offBuf, c0_w, bufA, 5, 2, 1);

  k_offconv_wmma<<<dim3(256, 7, NBATCH), blk32, 0, stream>>>(
      bufA, aStride, cs_off_w, cs_off_b, offBuf, 98, 7, 9, 3);
  k_deform<<<grdD, blkD, 0, stream>>>(bufA, aStride, offBuf, cs_w, bufB, 7, 9, 3);

  k_conv1x1_mul<<<dim3(256, 4, NBATCH), blk32, 0, stream>>>(
      bufB, conv2_w, conv2_b, x, out, 64);
}
